// LocalWindowAttention_88330297410145
// MI455X (gfx1250) — matrix operand order repaired and checked by reading, not compiled, not run on hardware
//
#include <hip/hip_runtime.h>

// MI455X / gfx1250: windowed + global-register attention.
// - All matmuls via v_wmma_f32_16x16x32_f16 (f16 in, f32 accumulate).
// - GEMMs double-buffer LDS tiles with global_load_async_to_lds_b128
//   (ASYNCcnt) so tile fill overlaps WMMA math; no VGPR staging.
// - f32->f16 conversion + weight transpose hoisted into tiny pre-passes.
//
// B=8, N=4096 (64x64 grid), HS=768, NH=12, HD=64, WS=8 (64-tok windows), NG=4.

typedef __attribute__((ext_vector_type(16))) _Float16 f16x16;
typedef __attribute__((ext_vector_type(8)))  _Float16 f16x8;
typedef __attribute__((ext_vector_type(4)))  _Float16 f16x4;
typedef __attribute__((ext_vector_type(8)))  float    fx8;

#define HSZ 768
#define QKV3 2304
#define NHEAD 12
#define HDIM 64
#define NTOK 4096
#define NBATCH 8

// ---------------------------------------------------------------------------
// CDNA5 async memory->LDS copy (16B per lane), tracked by ASYNCcnt.
// Generic pointers to __shared__ carry the LDS byte offset in their low 32
// bits (flat-aperture layout), which is exactly the dsaddr the instruction
// wants in VDST.
__device__ __forceinline__ unsigned lds_off(const void* p) {
    return (unsigned)(uintptr_t)p;
}
__device__ __forceinline__ void async_copy_b128(unsigned dst_lds, const void* src) {
    unsigned long long ga = (unsigned long long)(uintptr_t)src;
    asm volatile("global_load_async_to_lds_b128 %0, %1, off"
                 :: "v"(dst_lds), "v"(ga) : "memory");
}
__device__ __forceinline__ void wait_async0() {
    asm volatile("s_wait_asynccnt 0x0" ::: "memory");
}

// ---------------------------------------------------------------------------
// Fragment loaders (LDS f16, 16-byte-aligned rows).
// A-matrix 16x32: lane m=lane&15; halfs 0..15 -> K=16*(lane>>4)+j
// (contiguous, matching load_frag_b).
__device__ __forceinline__ f16x16 load_frag_a(const _Float16* base, int ld, int lane) {
    int m  = lane & 15;
    int kb = (lane >> 4) << 4;
    const _Float16* p = base + m * ld + kb;
    f16x16 f;
    ((f16x8*)&f)[0] = *(const f16x8*)(p);
    ((f16x8*)&f)[1] = *(const f16x8*)(p + 8);
    return f;
}
// B-matrix 32x16 (contiguous k, same as load_frag_a): lanes 0-15 hold K=0..15 contiguous,
// lanes 16-31 hold K=16..31; N = lane&15. Storage transposed [n][k].
__device__ __forceinline__ f16x16 load_frag_b(const _Float16* baseT, int ld, int lane) {
    int n  = lane & 15;
    int kb = (lane >> 4) << 4;
    const _Float16* p = baseT + n * ld + kb;
    f16x16 f;
    ((f16x8*)&f)[0] = *(const f16x8*)(p);
    ((f16x8*)&f)[1] = *(const f16x8*)(p + 8);
    return f;
}

// ---------------------------------------------------------------------------
// Pre-passes: f32 -> f16 copy, and f32 (K,N) -> f16 (N,K) transpose.
__global__ void cvt_f16_kernel(const float* __restrict__ s,
                               _Float16* __restrict__ d, int n4) {
    int i = blockIdx.x * 256 + threadIdx.x;
    if (i >= n4) return;
    float4 f = ((const float4*)s)[i];
    f16x4 h = { (_Float16)f.x, (_Float16)f.y, (_Float16)f.z, (_Float16)f.w };
    ((f16x4*)d)[i] = h;
}
__global__ void transpose_cvt_kernel(const float* __restrict__ s,
                                     _Float16* __restrict__ d, int K, int N) {
    int idx = blockIdx.x * 256 + threadIdx.x;   // over N*K, k contiguous
    if (idx >= K * N) return;
    int n = idx / K, k = idx - n * K;
    d[idx] = (_Float16)s[(size_t)k * N + n];    // coalesced f16 writes
}

// ---------------------------------------------------------------------------
// Pure-f16 WMMA GEMM: C(MxN) = A(MxK) @ Bt(NxK)^T.
// 128x128 block, BK=32, 256 threads = 8 waves (4M x 2N), wave tile 32x64.
// Double-buffered LDS filled by async copies; 1 barrier per K-step.
__device__ __forceinline__ void issue_tile(
    const _Float16* __restrict__ A, const _Float16* __restrict__ Bt,
    int K, int k0, unsigned ldsA, unsigned ldsB,
    int tid, int rowBase, int colBase) {
    #pragma unroll
    for (int j = 0; j < 2; ++j) {
        int i = tid + 256 * j;          // 16B chunk 0..511
        int r = i >> 2;                 // row 0..127
        int c = (i & 3) * 8;            // half col 0,8,16,24
        unsigned o = (unsigned)(r * 80 + c * 2);
        async_copy_b128(ldsA + o, A  + (size_t)(rowBase + r) * K + k0 + c);
        async_copy_b128(ldsB + o, Bt + (size_t)(colBase + r) * K + k0 + c);
    }
}

template <bool OUT_F16>
__global__ __launch_bounds__(256) void wmma_gemm_f16_kernel(
    const _Float16* __restrict__ A,    // (M, K) f16
    const _Float16* __restrict__ Bt,   // (N, K) f16 (transposed weights)
    _Float16* __restrict__ Ch, float* __restrict__ Cf,
    int K, int N) {
    __shared__ _Float16 Ah[2][128 * 40];   // [m][k], row stride 40 halfs
    __shared__ _Float16 Bh[2][128 * 40];   // [n][k]

    const int tid  = threadIdx.x;
    const int lane = tid & 31;
    const int wave = tid >> 5;
    const int rowBase = blockIdx.y * 128;
    const int colBase = blockIdx.x * 128;
    const int waveM = (wave >> 1) * 32;
    const int waveN = (wave & 1) * 64;

    const unsigned la[2] = { lds_off(&Ah[0][0]), lds_off(&Ah[1][0]) };
    const unsigned lb[2] = { lds_off(&Bh[0][0]), lds_off(&Bh[1][0]) };

    fx8 acc[2][4] = {};
    const int nk = K / 32;

    issue_tile(A, Bt, K, 0, la[0], lb[0], tid, rowBase, colBase);

    for (int ks = 0; ks < nk; ++ks) {
        const int cur = ks & 1;
        wait_async0();          // own async loads for buffer `cur` done
        __syncthreads();        // whole tile present; prev compute finished
        if (ks + 1 < nk)        // prefetch next tile behind this step's math
            issue_tile(A, Bt, K, (ks + 1) * 32, la[cur ^ 1], lb[cur ^ 1],
                       tid, rowBase, colBase);

        f16x16 af[2], bf[4];
        af[0] = load_frag_a(&Ah[cur][(waveM +  0) * 40], 40, lane);
        af[1] = load_frag_a(&Ah[cur][(waveM + 16) * 40], 40, lane);
        #pragma unroll
        for (int j = 0; j < 4; ++j)
            bf[j] = load_frag_b(&Bh[cur][(waveN + 16 * j) * 40], 40, lane);
        #pragma unroll
        for (int i = 0; i < 2; ++i)
            #pragma unroll
            for (int j = 0; j < 4; ++j)
                acc[i][j] = __builtin_amdgcn_wmma_f32_16x16x32_f16(
                    false, af[i], false, bf[j], (short)0, acc[i][j], false, false);
    }

    const int n = lane & 15, half = lane >> 4;
    #pragma unroll
    for (int i = 0; i < 2; ++i)
        #pragma unroll
        for (int j = 0; j < 4; ++j)
            #pragma unroll
            for (int v = 0; v < 8; ++v) {
                int row = rowBase + waveM + 16 * i + half * 8 + v;
                int col = colBase + waveN + 16 * j + n;
                if (OUT_F16) Ch[(size_t)row * N + col] = (_Float16)acc[i][j][v];
                else         Cf[(size_t)row * N + col] = acc[i][j][v];
            }
}

// ---------------------------------------------------------------------------
// Local window attention: one block per (head, window), 4 waves.
// RMS-norm q,k -> LDS f16; S = Q K^T and O = P V via WMMA; softmax f32.
// Epilogue fuses "+ og_mean[batch]" and writes f16 for the output GEMM.
__global__ __launch_bounds__(128) void local_attn_kernel(
    const _Float16* __restrict__ qkv,   // (32768, 2304) f16
    const float* __restrict__ qw, const float* __restrict__ kw,
    const float* __restrict__ og_mean,  // (8, 768) f32
    _Float16* __restrict__ oh)          // (32768, 768) f16
{
    __shared__ _Float16 Qs[64 * 72];
    __shared__ _Float16 Ks[64 * 72];
    __shared__ _Float16 Vt[64 * 72];        // [d][token]
    __shared__ _Float16 Ps[4][16 * 72];     // per-wave P strip

    const int h   = blockIdx.x;
    const int wid = blockIdx.y;
    const int b  = wid >> 6;
    const int w6 = wid & 63;
    const int wy = w6 >> 3, wx = w6 & 7;
    const int tid = threadIdx.x;

    auto tokRow = [&](int m) {
        return b * NTOK + (wy * 8 + (m >> 3)) * 64 + wx * 8 + (m & 7);
    };

    // ---- stage 1: rms-norm q (threads 0-63) / k (threads 64-127) ----
    {
        int m = tid & 63;
        bool isK = tid >= 64;
        const _Float16* src = qkv + (size_t)tokRow(m) * QKV3 + (isK ? HSZ : 0) + h * HDIM;
        const float* wn = isK ? kw : qw;
        _Float16* dst = (isK ? Ks : Qs) + m * 72;
        float vbuf[64];
        float ss = 0.f;
        #pragma unroll
        for (int d = 0; d < 64; ++d) { vbuf[d] = (float)src[d]; ss += vbuf[d] * vbuf[d]; }
        float r = rsqrtf(ss * (1.0f / 64.0f) + 1e-6f);
        #pragma unroll
        for (int d = 0; d < 64; ++d) dst[d] = (_Float16)(vbuf[d] * r * wn[d]);
    }
    // ---- V transposed into LDS ----
    {
        int m = tid >> 1;
        int dbase = (tid & 1) * 32;
        const _Float16* src = qkv + (size_t)tokRow(m) * QKV3 + 2 * HSZ + h * HDIM + dbase;
        #pragma unroll
        for (int j = 0; j < 32; ++j) Vt[(dbase + j) * 72 + m] = src[j];
    }
    __syncthreads();

    const int wave = tid >> 5, lane = tid & 31;
    const float scale = 0.125f;             // 64^-0.5

    // ---- S = Q K^T  (strip: 16 queries x 64 keys) ----
    fx8 s[4] = {};
    #pragma unroll
    for (int kk = 0; kk < 2; ++kk) {
        f16x16 aq = load_frag_a(Qs + (16 * wave) * 72 + kk * 32, 72, lane);
        #pragma unroll
        for (int j = 0; j < 4; ++j) {
            f16x16 bk = load_frag_b(Ks + (16 * j) * 72 + kk * 32, 72, lane);
            s[j] = __builtin_amdgcn_wmma_f32_16x16x32_f16(
                false, aq, false, bk, (short)0, s[j], false, false);
        }
    }

    // ---- softmax per row, write P (f16) to per-wave LDS ----
    const int n = lane & 15, half = lane >> 4;
    _Float16* Pw = &Ps[wave][0];
    #pragma unroll
    for (int v = 0; v < 8; ++v) {
        float x0 = s[0][v] * scale, x1 = s[1][v] * scale;
        float x2 = s[2][v] * scale, x3 = s[3][v] * scale;
        float mx = fmaxf(fmaxf(x0, x1), fmaxf(x2, x3));
        #pragma unroll
        for (int msk = 1; msk < 16; msk <<= 1) mx = fmaxf(mx, __shfl_xor(mx, msk, 32));
        float e0 = __expf(x0 - mx), e1 = __expf(x1 - mx);
        float e2 = __expf(x2 - mx), e3 = __expf(x3 - mx);
        float sm = e0 + e1 + e2 + e3;
        #pragma unroll
        for (int msk = 1; msk < 16; msk <<= 1) sm += __shfl_xor(sm, msk, 32);
        float inv = 1.0f / sm;
        int m = v + half * 8;
        Pw[m * 72 +  0 + n] = (_Float16)(e0 * inv);
        Pw[m * 72 + 16 + n] = (_Float16)(e1 * inv);
        Pw[m * 72 + 32 + n] = (_Float16)(e2 * inv);
        Pw[m * 72 + 48 + n] = (_Float16)(e3 * inv);
    }
    asm volatile("s_wait_dscnt 0x0" ::: "memory");  // cross-lane LDS round-trip

    // ---- O = P V ----
    fx8 ov[4] = {};
    #pragma unroll
    for (int kk = 0; kk < 2; ++kk) {
        f16x16 ap = load_frag_a(Pw + kk * 32, 72, lane);
        #pragma unroll
        for (int j = 0; j < 4; ++j) {
            f16x16 bv = load_frag_b(Vt + (16 * j) * 72 + kk * 32, 72, lane);
            ov[j] = __builtin_amdgcn_wmma_f32_16x16x32_f16(
                false, ap, false, bv, (short)0, ov[j], false, false);
        }
    }
    // ---- epilogue: + og_mean (per batch, per column), f16 out ----
    #pragma unroll
    for (int j = 0; j < 4; ++j) {
        int d = 16 * j + n;
        float bias = og_mean[(size_t)b * HSZ + h * HDIM + d];
        #pragma unroll
        for (int v = 0; v < 8; ++v) {
            int m = 16 * wave + v + half * 8;
            oh[(size_t)tokRow(m) * HSZ + h * HDIM + d] = (_Float16)(ov[j][v] + bias);
        }
    }
}

// ---------------------------------------------------------------------------
// Global-token QKV: gqkv(4 x 2304) = global_tokens(4 x 768) @ w_qkv (f32).
__global__ void global_qkv_kernel(const float* __restrict__ g,
                                  const float* __restrict__ wqkv,
                                  float* __restrict__ gqkv) {
    int e = blockIdx.x * 256 + threadIdx.x;
    if (e >= 4 * QKV3) return;
    int i = e / QKV3, j = e % QKV3;
    float acc = 0.f;
    for (int k = 0; k < HSZ; ++k) acc += g[i * HSZ + k] * wqkv[(size_t)k * QKV3 + j];
    gqkv[e] = acc;
}

// ---------------------------------------------------------------------------
// Global register-token attention (flash-style online softmax), one block per
// (head, batch); 4 queries over [4 global keys ; 4096 token keys]. Emits the
// q-mean directly: og_mean(B x 768).
__global__ __launch_bounds__(256) void global_attn_kernel(
    const _Float16* __restrict__ qkv, const float* __restrict__ gqkv,
    float* __restrict__ og_mean) {
    const int h = blockIdx.x, b = blockIdx.y;
    __shared__ float qg[4][64];
    __shared__ float sc[4][256];
    __shared__ float fsc[4], lsh[4], msh[4];
    __shared__ float ov[4][64];
    const int tid = threadIdx.x;
    const int q = tid >> 6, d = tid & 63;
    const float scale = 0.125f;

    qg[q][d] = gqkv[(size_t)q * QKV3 + h * HDIM + d];  // no qk_norm in global path
    if (tid < 4) { msh[tid] = -1e30f; lsh[tid] = 0.f; }
    float acc = 0.f;
    __syncthreads();

    for (int ch = 0; ch < 17; ++ch) {
        const int cnt = (ch == 0) ? 4 : 256;
        if (tid < cnt) {
            float kj[64];
            if (ch == 0) {
                #pragma unroll
                for (int dd = 0; dd < 64; ++dd)
                    kj[dd] = gqkv[(size_t)tid * QKV3 + HSZ + h * HDIM + dd];
            } else {
                int tok = (ch - 1) * 256 + tid;
                const _Float16* kp = qkv + (size_t)(b * NTOK + tok) * QKV3 + HSZ + h * HDIM;
                #pragma unroll
                for (int dd = 0; dd < 64; ++dd) kj[dd] = (float)kp[dd];
            }
            #pragma unroll
            for (int qq = 0; qq < 4; ++qq) {
                float sdot = 0.f;
                #pragma unroll
                for (int dd = 0; dd < 64; ++dd) sdot += qg[qq][dd] * kj[dd];
                sc[qq][tid] = sdot * scale;
            }
        }
        __syncthreads();
        if (tid < 4) {
            float cm = -1e30f;
            for (int j = 0; j < cnt; ++j) cm = fmaxf(cm, sc[tid][j]);
            float nm = fmaxf(msh[tid], cm);
            fsc[tid] = __expf(msh[tid] - nm);
            msh[tid] = nm;
        }
        __syncthreads();
        if (tid < cnt) {
            #pragma unroll
            for (int qq = 0; qq < 4; ++qq) sc[qq][tid] = __expf(sc[qq][tid] - msh[qq]);
        }
        __syncthreads();
        if (tid < 4) {
            float ssum = 0.f;
            for (int j = 0; j < cnt; ++j) ssum += sc[tid][j];
            lsh[tid] = lsh[tid] * fsc[tid] + ssum;
        }
        acc *= fsc[q];
        if (ch == 0) {
            for (int j = 0; j < 4; ++j)
                acc += sc[q][j] * gqkv[(size_t)j * QKV3 + 2 * HSZ + h * HDIM + d];
        } else {
            const _Float16* vp =
                qkv + (size_t)(b * NTOK + (ch - 1) * 256) * QKV3 + 2 * HSZ + h * HDIM + d;
            for (int j = 0; j < 256; ++j) acc += sc[q][j] * (float)vp[(size_t)j * QKV3];
        }
        __syncthreads();
    }

    ov[q][d] = acc / lsh[q];
    __syncthreads();
    if (tid < 64) {
        float m = 0.25f * (ov[0][tid] + ov[1][tid] + ov[2][tid] + ov[3][tid]);
        og_mean[(size_t)b * HSZ + h * HDIM + tid] = m;
    }
}

// ---------------------------------------------------------------------------
extern "C" void kernel_launch(void* const* d_in, const int* in_sizes, int n_in,
                              void* d_out, int out_size, void* d_ws, size_t ws_size,
                              hipStream_t stream) {
    const float* x      = (const float*)d_in[0];   // (8, 4096, 768)
    const float* w_qkv  = (const float*)d_in[1];   // (768, 2304)
    const float* w_out  = (const float*)d_in[2];   // (768, 768)
    const float* qnw    = (const float*)d_in[3];   // (64,)
    const float* knw    = (const float*)d_in[4];   // (64,)
    const float* gtok   = (const float*)d_in[5];   // (1, 4, 768)
    float* out = (float*)d_out;

    char* ws = (char*)d_ws;
    const size_t M = (size_t)NBATCH * NTOK;                 // 32768
    size_t off = 0;
    _Float16* qkvh = (_Float16*)(ws + off); off += M * QKV3 * sizeof(_Float16);      // 151.0 MB
    _Float16* oh   = (_Float16*)(ws + off); off += M * HSZ * sizeof(_Float16);       //  50.3 MB
    _Float16* xh   = (_Float16*)(ws + off); off += M * HSZ * sizeof(_Float16);       //  50.3 MB
    _Float16* wqT  = (_Float16*)(ws + off); off += (size_t)QKV3 * HSZ * sizeof(_Float16);
    _Float16* woT  = (_Float16*)(ws + off); off += (size_t)HSZ * HSZ * sizeof(_Float16);
    float* gqkv    = (float*)(ws + off);    off += 4 * QKV3 * sizeof(float);
    float* og_mean = (float*)(ws + off);

    // 0) precision pre-passes (f32 -> f16; weights transposed to (N,K))
    {
        int n4 = (int)(M * HSZ / 4);
        cvt_f16_kernel<<<(n4 + 255) / 256, 256, 0, stream>>>(x, xh, n4);
        int nq = HSZ * QKV3;
        transpose_cvt_kernel<<<(nq + 255) / 256, 256, 0, stream>>>(w_qkv, wqT, HSZ, QKV3);
        int no = HSZ * HSZ;
        transpose_cvt_kernel<<<(no + 255) / 256, 256, 0, stream>>>(w_out, woT, HSZ, HSZ);
    }
    // 1) global-token qkv (tiny, f32)
    global_qkv_kernel<<<(4 * QKV3 + 255) / 256, 256, 0, stream>>>(gtok, w_qkv, gqkv);
    // 2) QKV = x @ w_qkv -> f16 scratch      (M=32768, N=2304, K=768)
    wmma_gemm_f16_kernel<true><<<dim3(QKV3 / 128, M / 128), 256, 0, stream>>>(
        xh, wqT, qkvh, nullptr, HSZ, QKV3);
    // 3) global register attention -> og_mean
    global_attn_kernel<<<dim3(NHEAD, NBATCH), 256, 0, stream>>>(qkvh, gqkv, og_mean);
    // 4) local window attention (+og_mean fused) -> f16 oh
    local_attn_kernel<<<dim3(NHEAD, 512), 128, 0, stream>>>(qkvh, qnw, knw, og_mean, oh);
    // 5) out = oh @ w_out                    (M=32768, N=768, K=768)
    wmma_gemm_f16_kernel<false><<<dim3(HSZ / 128, M / 128), 256, 0, stream>>>(
        oh, woT, nullptr, out, HSZ, HSZ);
}
